// GATStockEncoder_3040836845573
// MI455X (gfx1250) — matrix operand order repaired and checked by reading, not compiled, not run on hardware
//
#include <hip/hip_runtime.h>
#include <hip/hip_bf16.h>

typedef _Float16 v16h __attribute__((ext_vector_type(16)));
typedef _Float16 v8h  __attribute__((ext_vector_type(8)));
typedef float    v8f  __attribute__((ext_vector_type(8)));

#define N_NODES   50000
#define N_EDGES   400000
#define E_TOT     (N_EDGES + N_NODES)   // self loops appended
#define D_IN      200
#define K_PAD     224                   // 7 * 32 WMMA K-steps (global pitch)
#define LDS_PITCH 232                   // padded LDS pitch (halves): 116 dwords -> bank-conflict free
#define HC        256                   // H * C
#define H_HEADS   4
#define C_DIM     64
#define BN_EPS    1e-5f
#define SLOPE     0.2f
#define M_ROWS_PAD 50048                // 391 * 128

// LDS layout for the GEMM kernel (bytes)
#define SA_HALVES (128 * LDS_PITCH)              // A tile: 128 rows
#define SB_HALVES (256 * LDS_PITCH)              // all of Wt: 256 rows
#define GEMM_LDS_BYTES ((SA_HALVES + SB_HALVES) * 2)

// ---------- monotone float<->uint encoding for atomic segment-max ----------
__device__ __forceinline__ unsigned enc_f32(float f) {
  unsigned u = __float_as_uint(f);
  return (u & 0x80000000u) ? ~u : (u | 0x80000000u);
}
__device__ __forceinline__ float dec_f32(unsigned k) {
  unsigned u = (k & 0x80000000u) ? (k ^ 0x80000000u) : ~k;
  return __uint_as_float(u);
}

// ---------- CDNA5 async global->LDS copy (16 B per lane), ASYNCcnt-tracked ----------
__device__ __forceinline__ void async_copy_b128(unsigned lds_byte_off, const void* gaddr) {
  asm volatile("global_load_async_to_lds_b128 %0, %1, off"
               :: "v"(lds_byte_off), "v"((unsigned long long)(size_t)gaddr)
               : "memory");
}
__device__ __forceinline__ void async_wait_all() {
  asm volatile("s_wait_asynccnt 0x0" ::: "memory");
}

// ---------- W1 [200,256] f32 row-major -> Wt [256,224] f16 (transposed, K-padded) ----------
__global__ void gat_repack_w(const float* __restrict__ W1, _Float16* __restrict__ Wt) {
  int t = blockIdx.x * blockDim.x + threadIdx.x;
  if (t >= HC * K_PAD) return;
  int n = t / K_PAD, k = t % K_PAD;
  float v = (k < D_IN) ? W1[(size_t)k * HC + n] : 0.0f;
  Wt[(size_t)n * K_PAD + k] = (_Float16)v;
}

// ---------- BN0 (eval) + cast to f16, zero K/M padding ----------
__global__ void gat_bn0_cast(const float* __restrict__ x,
                             const float* __restrict__ g, const float* __restrict__ b,
                             const float* __restrict__ m, const float* __restrict__ v,
                             _Float16* __restrict__ xh) {
  int n = blockIdx.x;          // 0 .. M_ROWS_PAD-1
  int k = threadIdx.x;         // 0 .. 223
  float o = 0.0f;
  if (n < N_NODES && k < D_IN) {
    float inv = rsqrtf(v[k] + BN_EPS);
    o = (x[(size_t)n * D_IN + k] - m[k]) * inv * g[k] + b[k];
  }
  xh[(size_t)n * K_PAD + k] = (_Float16)o;
}

// ---------- zero accumulators ----------
__global__ void gat_init(float* __restrict__ out, float* __restrict__ denom,
                         unsigned* __restrict__ menc) {
  size_t i = (size_t)blockIdx.x * blockDim.x + threadIdx.x;
  if (i < (size_t)N_NODES * HC) out[i] = 0.0f;
  if (i < (size_t)N_NODES * H_HEADS) { denom[i] = 0.0f; menc[i] = 0u; }
}

// ---------- GEMM: h = xh @ Wt^T via v_wmma_f32_16x16x32_f16 ----------
// Block: 256 threads (8 waves). LDS holds a 128-row A tile AND the whole Wt
// (both async-copied), so the inner loop is ds_load + wmma only.
// Each wave owns one 16-row M tile, loops all 16 N-tiles, K = 7 WMMA steps.
__global__ __launch_bounds__(256) void gat_gemm_wmma(const _Float16* __restrict__ xh,
                                                     const _Float16* __restrict__ Wt,
                                                     float* __restrict__ hbuf) {
  extern __shared__ __align__(32) _Float16 smem[];
  _Float16* sA = smem;                 // [128][LDS_PITCH]
  _Float16* sB = smem + SA_HALVES;     // [256][LDS_PITCH]

  const int tid = threadIdx.x;
  const size_t rowBase = (size_t)blockIdx.x * 128;

  // async cooperative A load: 128 rows * 28 chunks of 16 B (exact multiple of 256)
  for (int i = tid; i < 128 * (K_PAD / 8); i += 256) {
    int r = i / (K_PAD / 8);
    int c = i % (K_PAD / 8);
    async_copy_b128((unsigned)((r * LDS_PITCH + c * 8) * sizeof(_Float16)),
                    xh + (rowBase + r) * K_PAD + c * 8);
  }
  // async cooperative Wt load: 256 rows * 28 chunks of 16 B (exact multiple of 256)
  for (int i = tid; i < 256 * (K_PAD / 8); i += 256) {
    int r = i / (K_PAD / 8);
    int c = i % (K_PAD / 8);
    async_copy_b128((unsigned)((SA_HALVES + r * LDS_PITCH + c * 8) * sizeof(_Float16)),
                    Wt + (size_t)r * K_PAD + c * 8);
  }
  async_wait_all();
  __syncthreads();

  const int wave = tid >> 5;
  const int lane = tid & 31;
  const int lm   = lane & 15;   // M (A) / N (B,D) within tile
  const int hi   = lane >> 4;   // half-wave selector for K / M-rows of D

  // preload all A fragments for this wave's M tile into registers
  union Frag { v16h v; v8h h[2]; };
  const _Float16* aRow = &sA[(wave * 16 + lm) * LDS_PITCH];
  Frag afrag[K_PAD / 32];
#pragma unroll
  for (int kt = 0; kt < K_PAD / 32; ++kt) {
    afrag[kt].h[0] = *(const v8h*)(aRow + kt * 32 + hi * 8);
    afrag[kt].h[1] = *(const v8h*)(aRow + kt * 32 + 16 + hi * 8);
  }

  for (int nt = 0; nt < 16; ++nt) {
    v8f acc = {};
    const _Float16* bRow = &sB[(nt * 16 + lm) * LDS_PITCH];
#pragma unroll
    for (int kt = 0; kt < K_PAD / 32; ++kt) {
      Frag b;
      b.h[0] = *(const v8h*)(bRow + kt * 32 + hi * 8);
      b.h[1] = *(const v8h*)(bRow + kt * 32 + 16 + hi * 8);
      acc = __builtin_amdgcn_wmma_f32_16x16x32_f16(false, afrag[kt].v, false, b.v,
                                                   (short)0, acc, false, false);
    }
    // D layout: lane (lm,hi), VGPR r -> row 8*hi+r, col lm.
    // hbuf is padded to M_ROWS_PAD rows -> unconditional stores, no exec-mask ladder.
#pragma unroll
    for (int r = 0; r < 8; ++r) {
      size_t mrow = rowBase + wave * 16 + hi * 8 + r;
      hbuf[mrow * HC + nt * 16 + lm] = acc[r];
    }
  }
}

// ---------- a_src/a_dst: one wave per (node, head), lanes over channels ----------
__global__ void gat_att_dots(const float* __restrict__ hbuf,
                             const float* __restrict__ att_src,
                             const float* __restrict__ att_dst,
                             float* __restrict__ asrc, float* __restrict__ adst) {
  int wave = (blockIdx.x * blockDim.x + threadIdx.x) >> 5;
  int lane = threadIdx.x & 31;
  if (wave >= N_NODES * H_HEADS) return;
  int n = wave >> 2, h = wave & 3;
  const float* hp = hbuf + (size_t)n * HC + h * C_DIM;
  const float* as = att_src + h * C_DIM;
  const float* ad = att_dst + h * C_DIM;
  float s = 0.0f, d = 0.0f;
  for (int c = lane; c < C_DIM; c += 32) {
    float hv = hp[c];
    s += hv * as[c];
    d += hv * ad[c];
  }
  for (int off = 16; off > 0; off >>= 1) {
    s += __shfl_xor(s, off, 32);
    d += __shfl_xor(d, off, 32);
  }
  if (lane == 0) { asrc[wave] = s; adst[wave] = d; }
}

__device__ __forceinline__ void edge_src_dst(int e, const int* __restrict__ ei,
                                             int& src, int& dst) {
  if (e < N_EDGES) { src = ei[e]; dst = ei[N_EDGES + e]; }
  else             { src = dst = e - N_EDGES; }   // self loop
}

// ---------- edge pass 1: leaky-relu logits + atomic segment max ----------
__global__ void gat_edge_logits(const int* __restrict__ ei,
                                const float* __restrict__ asrc, const float* __restrict__ adst,
                                float* __restrict__ ebuf, unsigned* __restrict__ menc) {
  int t = blockIdx.x * blockDim.x + threadIdx.x;
  if (t >= E_TOT * H_HEADS) return;
  int e = t >> 2, h = t & 3;
  int src, dst; edge_src_dst(e, ei, src, dst);
  float v = asrc[src * H_HEADS + h] + adst[dst * H_HEADS + h];
  v = (v > 0.0f) ? v : SLOPE * v;
  ebuf[t] = v;
  atomicMax(&menc[dst * H_HEADS + h], enc_f32(v));
}

// ---------- edge pass 2: exp(e - max) + atomic segment sum ----------
__global__ void gat_edge_exp(const int* __restrict__ ei, float* __restrict__ ebuf,
                             const unsigned* __restrict__ menc, float* __restrict__ denom) {
  int t = blockIdx.x * blockDim.x + threadIdx.x;
  if (t >= E_TOT * H_HEADS) return;
  int e = t >> 2, h = t & 3;
  int src, dst; edge_src_dst(e, ei, src, dst);
  (void)src;
  float p = __expf(ebuf[t] - dec_f32(menc[dst * H_HEADS + h]));
  ebuf[t] = p;
  atomicAdd(&denom[dst * H_HEADS + h], p);
}

// ---------- edge pass 3: out[dst] += alpha * h[src]  (h is L2-resident) ----------
__global__ void gat_edge_aggregate(const int* __restrict__ ei,
                                   const float* __restrict__ ebuf,
                                   const float* __restrict__ denom,
                                   const float* __restrict__ hbuf,
                                   float* __restrict__ out) {
  int t = blockIdx.x * blockDim.x + threadIdx.x;
  if (t >= E_TOT * H_HEADS) return;
  int e = t >> 2, h = t & 3;
  int src, dst; edge_src_dst(e, ei, src, dst);
  float alpha = ebuf[t] / denom[dst * H_HEADS + h];
  const float4* hp = (const float4*)(hbuf + (size_t)src * HC + h * C_DIM);
  float* op = out + (size_t)dst * HC + h * C_DIM;
#pragma unroll 4
  for (int c4 = 0; c4 < C_DIM / 4; ++c4) {
    float4 hv = hp[c4];
    atomicAdd(&op[c4 * 4 + 0], alpha * hv.x);
    atomicAdd(&op[c4 * 4 + 1], alpha * hv.y);
    atomicAdd(&op[c4 * 4 + 2], alpha * hv.z);
    atomicAdd(&op[c4 * 4 + 3], alpha * hv.w);
  }
}

// ---------- epilogue: bias + ELU + BN1 (eval) ----------
__global__ void gat_epilogue(float* __restrict__ out, const float* __restrict__ bias,
                             const float* __restrict__ g, const float* __restrict__ b,
                             const float* __restrict__ m, const float* __restrict__ v) {
  size_t i = (size_t)blockIdx.x * blockDim.x + threadIdx.x;
  if (i >= (size_t)N_NODES * HC) return;
  int col = (int)(i & (HC - 1));
  float val = out[i] + bias[col];
  val = (val > 0.0f) ? val : (__expf(val) - 1.0f);
  float inv = rsqrtf(v[col] + BN_EPS);
  out[i] = (val - m[col]) * inv * g[col] + b[col];
}

extern "C" void kernel_launch(void* const* d_in, const int* in_sizes, int n_in,
                              void* d_out, int out_size, void* d_ws, size_t ws_size,
                              hipStream_t stream) {
  (void)in_sizes; (void)n_in; (void)out_size; (void)ws_size;
  const float* x       = (const float*)d_in[0];
  const int*   ei      = (const int*)  d_in[1];   // JAX default demotes int64 -> int32
  const float* bn0_g   = (const float*)d_in[2];
  const float* bn0_b   = (const float*)d_in[3];
  const float* bn0_m   = (const float*)d_in[4];
  const float* bn0_v   = (const float*)d_in[5];
  const float* W1      = (const float*)d_in[6];
  const float* att_src = (const float*)d_in[7];
  const float* att_dst = (const float*)d_in[8];
  const float* bias1   = (const float*)d_in[9];
  const float* bn1_g   = (const float*)d_in[10];
  const float* bn1_b   = (const float*)d_in[11];
  const float* bn1_m   = (const float*)d_in[12];
  const float* bn1_v   = (const float*)d_in[13];
  float* out = (float*)d_out;

  // workspace carve-out (256 B aligned slices)
  char* ws = (char*)d_ws;
  size_t off = 0;
  auto take = [&](size_t bytes) -> void* {
    void* p = ws + off;
    off = (off + bytes + 255) & ~(size_t)255;
    return p;
  };
  _Float16* xh    = (_Float16*)take((size_t)M_ROWS_PAD * K_PAD * sizeof(_Float16));
  _Float16* Wt    = (_Float16*)take((size_t)HC * K_PAD * sizeof(_Float16));
  float*    hbuf  = (float*)   take((size_t)M_ROWS_PAD * HC * sizeof(float));  // padded rows
  float*    asrc  = (float*)   take((size_t)N_NODES * H_HEADS * sizeof(float));
  float*    adst  = (float*)   take((size_t)N_NODES * H_HEADS * sizeof(float));
  unsigned* menc  = (unsigned*)take((size_t)N_NODES * H_HEADS * sizeof(unsigned));
  float*    denom = (float*)   take((size_t)N_NODES * H_HEADS * sizeof(float));
  float*    ebuf  = (float*)   take((size_t)E_TOT * H_HEADS * sizeof(float));

  gat_repack_w<<<(HC * K_PAD + 255) / 256, 256, 0, stream>>>(W1, Wt);
  gat_bn0_cast<<<M_ROWS_PAD, K_PAD, 0, stream>>>(x, bn0_g, bn0_b, bn0_m, bn0_v, xh);
  gat_init<<<((size_t)N_NODES * HC + 255) / 256, 256, 0, stream>>>(out, denom, menc);
  gat_gemm_wmma<<<(N_NODES + 127) / 128, 256, GEMM_LDS_BYTES, stream>>>(xh, Wt, hbuf);
  gat_att_dots<<<((size_t)N_NODES * H_HEADS * 32 + 255) / 256, 256, 0, stream>>>(
      hbuf, att_src, att_dst, asrc, adst);
  int eth = E_TOT * H_HEADS;
  gat_edge_logits<<<(eth + 255) / 256, 256, 0, stream>>>(ei, asrc, adst, ebuf, menc);
  gat_edge_exp<<<(eth + 255) / 256, 256, 0, stream>>>(ei, ebuf, menc, denom);
  gat_edge_aggregate<<<(eth + 255) / 256, 256, 0, stream>>>(ei, ebuf, denom, hbuf, out);
  gat_epilogue<<<((size_t)N_NODES * HC + 255) / 256, 256, 0, stream>>>(
      out, bias1, bn1_g, bn1_b, bn1_m, bn1_v);
}
